// Transformer_11871289606701
// MI455X (gfx1250) — matrix operand order repaired and checked by reading, not compiled, not run on hardware
//
#include <hip/hip_runtime.h>

typedef __attribute__((ext_vector_type(16))) _Float16 v16h;
typedef __attribute__((ext_vector_type(8)))  float    v8f;

#define WMMA_F16(a, b, c) \
  __builtin_amdgcn_wmma_f32_16x16x32_f16(false, (a), false, (b), (short)0, (c), false, false)

constexpr int Bb = 8;     // batch
constexpr int Cc = 512;   // channels
constexpr int Nn = 2048;  // sequence
constexpr int Dd = 64;    // head dim (C/8)

// Async 16-byte global->LDS copy (CDNA5 TDM-adjacent path, ASYNCcnt-tracked).
__device__ __forceinline__ void async_ld16(unsigned lds_off, const void* g) {
  asm volatile("global_load_async_to_lds_b128 %0, %1, off"
               :: "v"(lds_off), "v"(g) : "memory");
}
__device__ __forceinline__ void wait_async0() {
  asm volatile("s_wait_asynccnt 0" ::: "memory");
}

// ---------------------------------------------------------------------------
// Kernel 1: spectral sigma.  v = normalize(W^T u); sigma = ||Wv||^2/max(||Wv||,eps)
// ---------------------------------------------------------------------------
__global__ __launch_bounds__(256) void spectral_kernel(
    const float* __restrict__ W, const float* __restrict__ u,
    int O, int Cd, float* __restrict__ inv_sigma)
{
  __shared__ float red[256];
  __shared__ float vbuf[512];
  __shared__ float sh_scale;
  const int tid = threadIdx.x;

  float n2 = 0.f;
  for (int i = tid; i < Cd; i += 256) {
    float acc = 0.f;
    for (int o = 0; o < O; ++o) acc += W[(size_t)o * Cd + i] * u[o];
    vbuf[i] = acc;
    n2 += acc * acc;
  }
  red[tid] = n2; __syncthreads();
  for (int s = 128; s > 0; s >>= 1) { if (tid < s) red[tid] += red[tid + s]; __syncthreads(); }
  if (tid == 0) sh_scale = 1.f / fmaxf(sqrtf(red[0]), 1e-12f);
  __syncthreads();
  const float vs = sh_scale;

  float sn = 0.f;
  for (int o = tid; o < O; o += 256) {
    float acc = 0.f;
    for (int i = 0; i < Cd; ++i) acc += W[(size_t)o * Cd + i] * vbuf[i];
    acc *= vs;
    sn += acc * acc;
  }
  __syncthreads();
  red[tid] = sn; __syncthreads();
  for (int s = 128; s > 0; s >>= 1) { if (tid < s) red[tid] += red[tid + s]; __syncthreads(); }
  if (tid == 0) {
    float nn2 = red[0];
    float nn = sqrtf(nn2);
    float sigma = nn2 / fmaxf(nn, 1e-12f);
    inv_sigma[0] = 1.f / sigma;
  }
}

// ---------------------------------------------------------------------------
// Kernel 2: projection GEMM  out[o,n] = inv_sigma * sum_i W[o,i] x[b,i,n]
// f16 WMMA, f32 accumulate.  TRANS=1 -> out[N,Odim] (packed b128 stores),
// TRANS=0 -> out[Odim,N].
// ---------------------------------------------------------------------------
template <int TRANS>
__global__ __launch_bounds__(256) void proj_kernel(
    const float* __restrict__ W, const float* __restrict__ x,
    const float* __restrict__ inv_sigma, int sidx,
    _Float16* __restrict__ out, int Odim)
{
  __shared__ float xs[32][65];  // [k][n] tile, padded
  const int b = blockIdx.z, o0 = blockIdx.y * 64, n0 = blockIdx.x * 64;
  const int tid = threadIdx.x, w = tid >> 5, lane = tid & 31;
  const int l16 = lane & 15, lh = lane >> 4;
  const int o_sub = w & 3, nsub0 = (w >> 2);

  v8f c0, c1;
  #pragma unroll
  for (int i = 0; i < 8; ++i) { c0[i] = 0.f; c1[i] = 0.f; }

  const float* xb = x + (size_t)b * Cc * Nn;
  const int orow = o0 + o_sub * 16 + l16;

  for (int kb = 0; kb < Cc; kb += 32) {
    __syncthreads();
    for (int idx = tid; idx < 32 * 64; idx += 256) {
      int r = idx >> 6, cn = idx & 63;
      xs[r][cn] = xb[(size_t)(kb + r) * Nn + n0 + cn];
    }
    __syncthreads();

    const float* wr = W + (size_t)orow * Cc + kb + lh * 16;
    v16h a;
    #pragma unroll
    for (int j = 0; j < 8; ++j) { a[j] = (_Float16)wr[j]; a[8 + j] = (_Float16)wr[8 + j]; }

    const int ks = lh * 16;
    v16h b0, b1;
    #pragma unroll
    for (int j = 0; j < 16; ++j) {
      b0[j] = (_Float16)xs[ks + j][nsub0 * 16 + l16];
      b1[j] = (_Float16)xs[ks + j][(nsub0 + 2) * 16 + l16];
    }
    c0 = WMMA_F16(a, b0, c0);
    c1 = WMMA_F16(a, b1, c1);
  }

  const float scl = inv_sigma[sidx];
  if (TRANS) {
    // 8 rows per lane are contiguous in o: pack into one 16B store per tile
    union { uint4 u; _Float16 hh[8]; } p0, p1;
    #pragma unroll
    for (int v = 0; v < 8; ++v) {
      p0.hh[v] = (_Float16)(c0[v] * scl);
      p1.hh[v] = (_Float16)(c1[v] * scl);
    }
    const int o_base = o0 + o_sub * 16 + lh * 8;
    const int n0a = n0 + nsub0 * 16 + l16;
    const int n1a = n0 + (nsub0 + 2) * 16 + l16;
    *(uint4*)&out[((size_t)b * Nn + n0a) * Odim + o_base] = p0.u;
    *(uint4*)&out[((size_t)b * Nn + n1a) * Odim + o_base] = p1.u;
  } else {
    #pragma unroll
    for (int v = 0; v < 8; ++v) {
      const int o_abs = o0 + o_sub * 16 + lh * 8 + v;
      const int n0a = n0 + nsub0 * 16 + l16;
      const int n1a = n0 + (nsub0 + 2) * 16 + l16;
      out[((size_t)b * Odim + o_abs) * Nn + n0a] = (_Float16)(c0[v] * scl);
      out[((size_t)b * Odim + o_abs) * Nn + n1a] = (_Float16)(c1[v] * scl);
    }
  }
}

// ---------------------------------------------------------------------------
// Kernel 3: fused flash attention (softmax over keys n, query index m).
// Grid (N/128, B), 256 threads (8 waves). Wave w owns query subtile
// m0 = 128*blockIdx.x + 16*w, all 512 channels (32 accumulator tiles).
// h tiles double-buffered in LDS via async global->LDS copies (ASYNCcnt).
// ---------------------------------------------------------------------------
__global__ __launch_bounds__(256) void attn_kernel(
    const _Float16* __restrict__ fT, const _Float16* __restrict__ gT,
    const _Float16* __restrict__ hM, const float* __restrict__ x,
    const float* __restrict__ gamma, float* __restrict__ out)
{
  __shared__ _Float16 hs[2][Cc * 32];  // double-buffered h tile [c][n] (2x32 KB)
  __shared__ _Float16 ps[8][32 * 16];  // per-wave P tile [n][m] (8 KB)

  const int b = blockIdx.y, mt = blockIdx.x;
  const int tid = threadIdx.x, w = tid >> 5, lane = tid & 31;
  const int l16 = lane & 15, lh = lane >> 4;
  const int m0 = mt * 128 + w * 16;

  const _Float16* fTb = fT + (size_t)b * Nn * Dd;
  const _Float16* gTb = gT + (size_t)b * Nn * Dd;
  const _Float16* hb  = hM + (size_t)b * Cc * Nn;

  // stage: issue 2048 async b128 copies of h[:, nb..nb+31] into hs[buf]
  auto stage = [&](int nb, int buf) {
    const unsigned base = (unsigned)(uintptr_t)&hs[buf][0];
    for (int idx = tid; idx < 2048; idx += 256) {
      const int cpos = idx >> 2, ch = idx & 3;
      const unsigned lds_off = base + (unsigned)((cpos * 32 + ch * 8) * 2);
      async_ld16(lds_off, &hb[(size_t)cpos * Nn + nb + ch * 8]);
    }
  };

  // Preload query B-operands (fixed per wave): column m over d (K=0..31, 32..63)
  v16h bq0, bq1;
  {
    const _Float16* gr = gTb + (size_t)(m0 + l16) * Dd + lh * 8;
    #pragma unroll
    for (int j = 0; j < 8; ++j) { bq0[j] = gr[j]; bq0[8 + j] = gr[16 + j]; bq1[j] = gr[32 + j]; bq1[8 + j] = gr[48 + j]; }
  }

  v8f acc[32];
  #pragma unroll
  for (int t = 0; t < 32; ++t) {
    #pragma unroll
    for (int v = 0; v < 8; ++v) acc[t][v] = 0.f;
  }
  float run_max = -1e30f, run_sum = 0.f;
  _Float16* pw = &ps[w][0];

  stage(0, 0);
  int buf = 0;
  for (int nb = 0; nb < Nn; nb += 32) {
    wait_async0();      // own async copies done
    __syncthreads();    // everyone's copies visible
    if (nb + 32 < Nn) stage(nb + 32, buf ^ 1);  // prefetch next tile
    const _Float16* hcur = &hs[buf][0];

    // ---- S = f^T g for n-subtiles [nb..nb+15], [nb+16..nb+31]
    v8f s0, s1;
    #pragma unroll
    for (int i = 0; i < 8; ++i) { s0[i] = 0.f; s1[i] = 0.f; }
    {
      const _Float16* fr = fTb + (size_t)(nb + l16) * Dd + lh * 8;
      v16h a0, a1;
      #pragma unroll
      for (int j = 0; j < 8; ++j) {
        a0[j] = fr[j];      a0[8 + j] = fr[16 + j];
        a1[j] = fr[32 + j]; a1[8 + j] = fr[48 + j];
      }
      s0 = WMMA_F16(a0, bq0, s0);
      s0 = WMMA_F16(a1, bq1, s0);
    }
    {
      const _Float16* fr = fTb + (size_t)(nb + 16 + l16) * Dd + lh * 8;
      v16h a0, a1;
      #pragma unroll
      for (int j = 0; j < 8; ++j) {
        a0[j] = fr[j];      a0[8 + j] = fr[16 + j];
        a1[j] = fr[32 + j]; a1[8 + j] = fr[48 + j];
      }
      s1 = WMMA_F16(a0, bq0, s1);
      s1 = WMMA_F16(a1, bq1, s1);
    }

    // ---- online softmax over n (column m lives in lanes {l16, l16+16})
    float tmax = -1e30f;
    #pragma unroll
    for (int v = 0; v < 8; ++v) tmax = fmaxf(tmax, fmaxf(s0[v], s1[v]));
    tmax = fmaxf(tmax, __shfl_xor(tmax, 16, 32));
    const float new_max = fmaxf(run_max, tmax);
    const float alpha = __expf(run_max - new_max);
    float tsum = 0.f;
    #pragma unroll
    for (int v = 0; v < 8; ++v) {
      s0[v] = __expf(s0[v] - new_max);
      s1[v] = __expf(s1[v] - new_max);
      tsum += s0[v] + s1[v];
    }
    tsum += __shfl_xor(tsum, 16, 32);
    run_sum = run_sum * alpha + tsum;
    run_max = new_max;

    #pragma unroll
    for (int t = 0; t < 32; ++t) {
      #pragma unroll
      for (int v = 0; v < 8; ++v) acc[t][v] *= alpha;
    }

    // ---- stage P[n][m] to LDS (D layout row = vgpr + 8*lh), re-read as B op
    #pragma unroll
    for (int v = 0; v < 8; ++v) {
      pw[(lh * 8 + v) * 16 + l16]      = (_Float16)s0[v];
      pw[(16 + lh * 8 + v) * 16 + l16] = (_Float16)s1[v];
    }
    __builtin_amdgcn_wave_barrier();  // per-wave LDS RAW; DS ops are in-order
    v16h bp;
    #pragma unroll
    for (int j = 0; j < 16; ++j) bp[j] = pw[(j < 8 ? lh * 8 + j : 8 + lh * 8 + j) * 16 + l16];

    // ---- O[c,m] += h[c,nb..nb+31] * P   (32 channel tiles)
    #pragma unroll
    for (int ct = 0; ct < 32; ++ct) {
      const _Float16* hr = &hcur[(ct * 16 + l16) * 32 + lh * 8];
      v16h a;
      #pragma unroll
      for (int j = 0; j < 8; ++j) { a[j] = hr[j]; a[8 + j] = hr[16 + j]; }
      acc[ct] = WMMA_F16(a, bp, acc[ct]);
    }
    buf ^= 1;
  }

  // ---- epilogue: out = gamma * acc / l + x
  const float inv_l = 1.f / run_sum;
  const float gm = gamma[0];
  const float* xb = x + (size_t)b * Cc * Nn;
  float* ob = out + (size_t)b * Cc * Nn;
  #pragma unroll
  for (int ct = 0; ct < 32; ++ct) {
    #pragma unroll
    for (int v = 0; v < 8; ++v) {
      const int c_abs = ct * 16 + lh * 8 + v;
      const size_t off = (size_t)c_abs * Nn + (m0 + l16);
      ob[off] = gm * acc[ct][v] * inv_l + xb[off];
    }
  }
}

// ---------------------------------------------------------------------------
extern "C" void kernel_launch(void* const* d_in, const int* in_sizes, int n_in,
                              void* d_out, int out_size, void* d_ws, size_t ws_size,
                              hipStream_t stream)
{
  const float* x     = (const float*)d_in[0];
  const float* Wq    = (const float*)d_in[1];
  const float* Wk    = (const float*)d_in[2];
  const float* Wv    = (const float*)d_in[3];
  const float* uq    = (const float*)d_in[4];
  const float* uk    = (const float*)d_in[5];
  const float* uv    = (const float*)d_in[6];
  const float* gamma = (const float*)d_in[7];
  float* out = (float*)d_out;

  char* ws = (char*)d_ws;
  float*    inv_sigma = (float*)ws;                    // 3 floats (+pad)
  _Float16* fT = (_Float16*)(ws + 64);                 // [B,N,64] f16 (2 MB)
  _Float16* gT = fT + (size_t)Bb * Nn * Dd;            // [B,N,64] f16 (2 MB)
  _Float16* hM = gT + (size_t)Bb * Nn * Dd;            // [B,C,N] f16 (16 MB)

  spectral_kernel<<<dim3(1), dim3(256), 0, stream>>>(Wq, uq, Dd, Cc, inv_sigma + 0);
  spectral_kernel<<<dim3(1), dim3(256), 0, stream>>>(Wk, uk, Dd, Cc, inv_sigma + 1);
  spectral_kernel<<<dim3(1), dim3(256), 0, stream>>>(Wv, uv, Cc, Cc, inv_sigma + 2);

  proj_kernel<1><<<dim3(Nn / 64, Dd / 64, Bb), dim3(256), 0, stream>>>(
      Wq, x, inv_sigma, 0, fT, Dd);
  proj_kernel<1><<<dim3(Nn / 64, Dd / 64, Bb), dim3(256), 0, stream>>>(
      Wk, x, inv_sigma, 1, gT, Dd);
  proj_kernel<0><<<dim3(Nn / 64, Cc / 64, Bb), dim3(256), 0, stream>>>(
      Wv, x, inv_sigma, 2, hM, Cc);

  attn_kernel<<<dim3(Nn / 128, Bb), dim3(256), 0, stream>>>(
      fT, gT, hM, x, gamma, out);
}
